// BinaryClassifier_1735166788531
// MI455X (gfx1250) — matrix operand order repaired and checked by reading, not compiled, not run on hardware
//
#include <hip/hip_runtime.h>
#include <cstdint>
#include <cstddef>

// ---------------- problem dims ----------------
#define NN   20000      // nodes
#define NE   200000     // edges
#define NIN  64         // node in feats
#define EIN  16         // edge in feats
#define HIDW 64         // hidden per head
#define NH   4          // heads
#define FD   256        // NH*HIDW
#define NT   512        // trip nodes
#define NPAIR (512*512)

typedef __attribute__((ext_vector_type(16))) _Float16 v16h;
typedef __attribute__((ext_vector_type(8)))  float    v8f;

__device__ __forceinline__ v8f wmma16(const v16h a, const v16h b, const v8f c) {
  // D(16x16 f32) = A(16x32 f16) x B(32x16 f16) + C
  return __builtin_amdgcn_wmma_f32_16x16x32_f16(false, a, false, b, (short)0, c, false, false);
}

// A fragment (16x32, row-major source, lda elements): lane m = lane%16,
// K-halves interleaved per ISA 7.12.2 (VGPR v holds K={2v,2v+1}+8*(lane>=16), +16 for v>=4)
template<typename T, bool GUARD>
__device__ __forceinline__ v16h load_a_frag(const T* A, int lda, int k0, int kreal) {
  const int lane = threadIdx.x & 31;
  const int m    = lane & 15;
  const int koff = (lane >> 4) << 3;           // 0 or 8
  v16h a;
#pragma unroll
  for (int p = 0; p < 8; ++p) {
    const int k = k0 + ((p & 4) << 2) + koff + ((p & 3) << 1);
    float x0 = 0.f, x1 = 0.f;
    if (!GUARD || (k     < kreal)) x0 = (float)A[m * lda + k];
    if (!GUARD || (k + 1 < kreal)) x1 = (float)A[m * lda + k + 1];
    a[2 * p]     = (_Float16)x0;
    a[2 * p + 1] = (_Float16)x1;
  }
  return a;
}

// B fragment (32x16) from pre-transposed weights Wt[n][k] (ldk = padded K):
// lane n = lane%16, K = k0 + 8*(lane>=16) + e (e<8), k0 + 16 + 8*(lane>=16) + (e-8) (e>=8)
__device__ __forceinline__ v16h load_b_frag(const _Float16* Wt, int ldk, int n0, int k0) {
  const int lane = threadIdx.x & 31;
  const _Float16* p = Wt + (size_t)(n0 + (lane & 15)) * ldk + (k0 + ((lane >> 4) << 3));
  v16h b;
#pragma unroll
  for (int e = 0; e < 16; ++e) b[e] = p[e + ((e >> 3) << 3)];
  return b;
}

// order-preserving float<->uint for atomicMax-based segment max
__device__ __forceinline__ unsigned fenc(float f) {
  unsigned u = __float_as_uint(f);
  return (u & 0x80000000u) ? ~u : (u | 0x80000000u);
}
__device__ __forceinline__ float fdec(unsigned k) {
  return (k & 0x80000000u) ? __uint_as_float(k & 0x7FFFFFFFu) : __uint_as_float(~k);
}

// ---------------- weight prep: f32 -> f16, transposed, K zero-padded ----------------
__global__ void prep_wt_kernel(const float* __restrict__ W, _Float16* __restrict__ Wt,
                               int Kr, int NC, int Kpad, int rowOff) {
  const int t = blockIdx.x * 256 + threadIdx.x;
  if (t >= NC * Kpad) return;
  const int n = t / Kpad, k = t - n * Kpad;
  Wt[t] = (k < Kr) ? (_Float16)W[(size_t)(rowOff + k) * NC + n] : (_Float16)0.f;
}

// ---------------- generic GEMM: Out(f16, MxNC) = A(MxK) @ W(KxNC) (+bias) ----------------
template<typename TA, bool GUARD>
__global__ void gemm_kernel(const TA* __restrict__ A, int lda, int Kreal, int Kpad,
                            const _Float16* __restrict__ Wt, const float* __restrict__ bias,
                            _Float16* __restrict__ Out, int ldo, int M) {
  const int wave = threadIdx.x >> 5;
  const int row0 = (blockIdx.x * (blockDim.x >> 5) + wave) * 16;
  if (row0 >= M) return;
  const int col0 = blockIdx.y * 64;
  const TA* Arow = A + (size_t)row0 * lda;
  v8f c[4] = {};
  for (int k0 = 0; k0 < Kpad; k0 += 32) {
    if (k0 + 32 < Kpad) __builtin_prefetch(Arow + (k0 + 32), 0, 1);
    v16h a = load_a_frag<TA, GUARD>(Arow, lda, k0, Kreal);
#pragma unroll
    for (int t = 0; t < 4; ++t) {
      v16h b = load_b_frag(Wt, Kpad, col0 + t * 16, k0);
      c[t] = wmma16(a, b, c[t]);
    }
  }
  const int lane = threadIdx.x & 31;
  const int n = lane & 15, mb = (lane >> 4) << 3;
#pragma unroll
  for (int t = 0; t < 4; ++t) {
    const int cc = col0 + t * 16 + n;
    const float bv = bias ? bias[cc] : 0.f;
#pragma unroll
    for (int r = 0; r < 8; ++r)
      Out[(size_t)(row0 + mb + r) * ldo + cc] = (_Float16)(c[t][r] + bv);
  }
}

// ---------------- fused EGAT edge kernel ----------------
// f_fij = ef @ Wfij (WMMA); f_tmp = f_fij + fni[src] + fnj[dst]; leaky_relu;
// optional f16 store of f_out (next-layer ef, in place, but racy: the 4 head blocks sharing rows read all cols);
// attention logits e[edge][head] = sum_hid f_out * attn (cross-lane shfl reduction).
template<typename TA, bool GUARD, bool WRITE_EOUT>
__global__ void edge_kernel(const TA* EF, int lda, int Kreal, int Kpad,
                            const _Float16* __restrict__ Wt,
                            const _Float16* __restrict__ fni, const _Float16* __restrict__ fnj,
                            const int* __restrict__ src, const int* __restrict__ dst,
                            const float* __restrict__ attn,
                            _Float16* EOut, float* __restrict__ logits, int E_) {
  const int wave = threadIdx.x >> 5;
  const int row0 = (blockIdx.x * (blockDim.x >> 5) + wave) * 16;
  if (row0 >= E_) return;
  const int head = blockIdx.y;
  const int col0 = head * 64;
  const TA* Arow = EF + (size_t)row0 * lda;
  v8f c[4] = {};
  for (int k0 = 0; k0 < Kpad; k0 += 32) {
    if (k0 + 32 < Kpad) __builtin_prefetch(Arow + (k0 + 32), 0, 1);
    v16h a = load_a_frag<TA, GUARD>(Arow, lda, k0, Kreal);
#pragma unroll
    for (int t = 0; t < 4; ++t) {
      v16h b = load_b_frag(Wt, Kpad, col0 + t * 16, k0);
      c[t] = wmma16(a, b, c[t]);
    }
  }
  const int lane = threadIdx.x & 31;
  const int n = lane & 15, mb = (lane >> 4) << 3;
  float part[8] = {};
#pragma unroll
  for (int t = 0; t < 4; ++t) {
    const int cc = col0 + t * 16 + n;
    const float aw = attn[cc];
#pragma unroll
    for (int r = 0; r < 8; ++r) {
      const int row = row0 + mb + r;
      const int s = src[row], d = dst[row];
      float v = c[t][r] + (float)fni[(size_t)s * FD + cc] + (float)fnj[(size_t)d * FD + cc];
      v = v > 0.f ? v : 0.01f * v;                  // leaky_relu
      if (WRITE_EOUT) EOut[(size_t)row * FD + cc] = (_Float16)v;
      part[r] += v * aw;
    }
  }
#pragma unroll
  for (int off = 8; off >= 1; off >>= 1)
#pragma unroll
    for (int r = 0; r < 8; ++r) part[r] += __shfl_xor(part[r], off, 32);
  if (n == 0) {
#pragma unroll
    for (int r = 0; r < 8; ++r) logits[(size_t)(row0 + mb + r) * NH + head] = part[r];
  }
}

// ---------------- edge softmax + aggregation ----------------
__global__ void segmax_kernel(const float* __restrict__ logits, const int* __restrict__ dst,
                              unsigned* __restrict__ menc, int E_) {
  const int t = blockIdx.x * 256 + threadIdx.x;
  if (t >= E_ * NH) return;
  const int e = t >> 2, h = t & 3;
  atomicMax(&menc[(size_t)dst[e] * NH + h], fenc(logits[t]));
}

__global__ void expsum_kernel(float* __restrict__ logits, const int* __restrict__ dst,
                              const unsigned* __restrict__ menc, float* __restrict__ ssum, int E_) {
  const int t = blockIdx.x * 256 + threadIdx.x;
  if (t >= E_ * NH) return;
  const int e = t >> 2, h = t & 3;
  const int d = dst[e];
  const float ee = __expf(logits[t] - fdec(menc[(size_t)d * NH + h]));
  logits[t] = ee;
  atomicAdd(&ssum[(size_t)d * NH + h], ee);
}

__global__ void msg_kernel(const _Float16* __restrict__ hnode, const float* __restrict__ ee,
                           const float* __restrict__ ssum, const int* __restrict__ src,
                           const int* __restrict__ dst, float* __restrict__ hout, long total) {
  const long t = (long)blockIdx.x * 256 + threadIdx.x;
  if (t >= total) return;                         // total = E*32, 8 cols per thread
  const int e  = (int)(t >> 5);
  const int c0 = (int)(t & 31) << 3;
  const int s = src[e], d = dst[e];
  const int h = c0 >> 6;
  const float a = ee[(size_t)e * NH + h] / ssum[(size_t)d * NH + h];
  const _Float16* hp = hnode + (size_t)s * FD + c0;
  float* op = hout + (size_t)d * FD + c0;
#pragma unroll
  for (int j = 0; j < 8; ++j) atomicAdd(op + j, (float)hp[j] * a);
}

__global__ void f2h_kernel(const float* __restrict__ in, _Float16* __restrict__ out, long n) {
  const long t = (long)blockIdx.x * 256 + threadIdx.x;
  if (t < n) out[t] = (_Float16)in[t];
}

// ---------------- head: trip gather + MLPs ----------------
__global__ void trip_kernel(const int* __restrict__ mask, int* __restrict__ tidx, int Nn, int ntrip) {
  if (threadIdx.x == 0 && blockIdx.x == 0) {
    int c = 0;
    for (int i = 0; i < Nn && c < ntrip; ++i) if (mask[i]) tidx[c++] = i;
    for (; c < ntrip; ++c) tidx[c] = 0;           // jnp.nonzero(size=) pads with 0
  }
}

__global__ void gather_kernel(const float* __restrict__ hout, const float* __restrict__ nfeat,
                              const int* __restrict__ tidx, _Float16* __restrict__ htrip) {
  const int t = blockIdx.x * 256 + threadIdx.x;
  if (t >= NT * 128) return;
  const int row = t >> 7, c = t & 127;
  const int node = tidx[row];
  float v;
  if (c < HIDW) {                                  // nf3 = mean over heads
    const float* hp = hout + (size_t)node * FD + c;
    v = 0.25f * (hp[0] + hp[64] + hp[128] + hp[192]);
  } else {
    v = nfeat[(size_t)node * NIN + (c - HIDW)];
  }
  htrip[t] = (_Float16)v;
}

// one wave: 16 rows through one dense layer; act LDS lda fixed = 128
__device__ __forceinline__ void mlp_layer(const _Float16* actIn, _Float16* actOut,
                                          const _Float16* Wt, const float* bias,
                                          int K, int NC, bool relu) {
  const int lane = threadIdx.x & 31;
  const int n = lane & 15, mb = (lane >> 4) << 3;
  for (int nt = 0; nt < (NC >> 4); ++nt) {
    v8f c = {};
    for (int k0 = 0; k0 < K; k0 += 32) {
      v16h a = load_a_frag<_Float16, false>(actIn, 128, k0, K);
      v16h b = load_b_frag(Wt, K, nt * 16, k0);
      c = wmma16(a, b, c);
    }
    const float bv = bias[nt * 16 + n];
#pragma unroll
    for (int r = 0; r < 8; ++r) {
      float v = c[r] + bv;
      if (relu) v = v > 0.f ? v : 0.f;
      actOut[(mb + r) * 128 + nt * 16 + n] = (_Float16)v;
    }
  }
}

__global__ void ml_kernel(const _Float16* __restrict__ htrip,
                          const _Float16* __restrict__ w0, const float* __restrict__ b0,
                          const _Float16* __restrict__ w1, const float* __restrict__ b1,
                          const _Float16* __restrict__ w2, const float* __restrict__ b2,
                          const _Float16* __restrict__ w3, const float* __restrict__ b3,
                          _Float16* __restrict__ hfin) {
  __shared__ _Float16 sA[4][16 * 128];
  __shared__ _Float16 sB[4][16 * 128];
  const int wave = threadIdx.x >> 5, lane = threadIdx.x & 31;
  const int row0 = (blockIdx.x * 4 + wave) * 16;   // 8 blocks * 4 waves * 16 = 512
  const _Float16* g = htrip + (size_t)row0 * 128;
  for (int t = lane; t < 16 * 128; t += 32) sA[wave][t] = g[t];
  mlp_layer(sA[wave], sB[wave], w0, b0, 128, 128, true);
  mlp_layer(sB[wave], sA[wave], w1, b1, 128, 128, true);
  mlp_layer(sA[wave], sB[wave], w2, b2, 128, 128, true);
  mlp_layer(sB[wave], sA[wave], w3, b3, 128, 64, false);
  _Float16* o = hfin + (size_t)row0 * 64;
  for (int t = lane; t < 16 * 64; t += 32) {
    const int r = t >> 6, cc = t & 63;
    o[t] = sA[wave][r * 128 + cc];
  }
}

// U = hfin @ W0[:64], V = hfin @ W0[64:]  (pair-head first layer split)
__global__ void uv_kernel(const _Float16* __restrict__ hf,
                          const _Float16* __restrict__ wtT, const _Float16* __restrict__ wtB,
                          float* __restrict__ U, float* __restrict__ V) {
  const int wave = threadIdx.x >> 5;
  const int row0 = (blockIdx.x * 4 + wave) * 16;
  const _Float16* A = hf + (size_t)row0 * 64;
  v8f cu[2] = {}, cv[2] = {};
  for (int k0 = 0; k0 < 64; k0 += 32) {
    v16h a = load_a_frag<_Float16, false>(A, 64, k0, 64);
#pragma unroll
    for (int t = 0; t < 2; ++t) {
      cu[t] = wmma16(a, load_b_frag(wtT, 64, t * 16, k0), cu[t]);
      cv[t] = wmma16(a, load_b_frag(wtB, 64, t * 16, k0), cv[t]);
    }
  }
  const int lane = threadIdx.x & 31;
  const int n = lane & 15, mb = (lane >> 4) << 3;
#pragma unroll
  for (int t = 0; t < 2; ++t)
#pragma unroll
    for (int r = 0; r < 8; ++r) {
      U[(size_t)(row0 + mb + r) * 32 + t * 16 + n] = cu[t][r];
      V[(size_t)(row0 + mb + r) * 32 + t * 16 + n] = cv[t][r];
    }
}

// fused all-pairs MLP: act0 = relu(U[i]+V[j]+b0), then 6 WMMA layers, 16->1 dot + sigmoid
__global__ void pairs_kernel(const float* __restrict__ U, const float* __restrict__ V,
                             const float* __restrict__ b0,
                             const _Float16* __restrict__ w1, const float* __restrict__ b1,
                             const _Float16* __restrict__ w2, const float* __restrict__ b2,
                             const _Float16* __restrict__ w3, const float* __restrict__ b3,
                             const _Float16* __restrict__ w4, const float* __restrict__ b4,
                             const _Float16* __restrict__ w5, const float* __restrict__ b5,
                             const _Float16* __restrict__ w6, const float* __restrict__ b6,
                             const float* __restrict__ w7, const float* __restrict__ b7,
                             float* __restrict__ out) {
  __shared__ _Float16 sA[4][16 * 128];
  __shared__ _Float16 sB[4][16 * 128];
  const int wave = threadIdx.x >> 5, lane = threadIdx.x & 31;
  const long pbase = ((long)blockIdx.x * 4 + wave) * 16;
  // act0: 16 rows x 32 cols
  for (int t = lane; t < 16 * 32; t += 32) {
    const int r = t >> 5, cc = t & 31;
    const long p = pbase + r;
    const int i = (int)(p >> 9), j = (int)(p & 511);
    float v = U[(size_t)i * 32 + cc] + V[(size_t)j * 32 + cc] + b0[cc];
    sA[wave][r * 128 + cc] = (_Float16)(v > 0.f ? v : 0.f);
  }
  mlp_layer(sA[wave], sB[wave], w1, b1, 32, 64, true);
  mlp_layer(sB[wave], sA[wave], w2, b2, 64, 128, true);
  mlp_layer(sA[wave], sB[wave], w3, b3, 128, 128, true);
  mlp_layer(sB[wave], sA[wave], w4, b4, 128, 64, true);
  mlp_layer(sA[wave], sB[wave], w5, b5, 64, 32, true);
  mlp_layer(sB[wave], sA[wave], w6, b6, 32, 16, true);
  if (lane < 16) {
    const _Float16* ai = sA[wave] + lane * 128;
    float acc = b7[0];
#pragma unroll
    for (int k = 0; k < 16; ++k) acc += (float)ai[k] * w7[k];
    out[pbase + lane] = 1.f / (1.f + __expf(-acc));
  }
}

// ---------------- host ----------------
extern "C" void kernel_launch(void* const* d_in, const int* in_sizes, int n_in,
                              void* d_out, int out_size, void* d_ws, size_t ws_size,
                              hipStream_t stream) {
  (void)in_sizes; (void)n_in; (void)out_size; (void)ws_size;
  // input order: nodes_feats, edges_feats, src, dst, mask, then params pytree
  // (sorted dict keys): egat0..2{Wfij,Wni,Wnj,Wnode,attn,bnode}, l[0..7]{W,b}, ml[0..3]{W,b}
  const float* nfeat = (const float*)d_in[0];
  const float* efeat = (const float*)d_in[1];
  const int*   src   = (const int*)d_in[2];
  const int*   dst   = (const int*)d_in[3];
  const int*   mask  = (const int*)d_in[4];
  const float *eg_fij[3], *eg_ni[3], *eg_nj[3], *eg_nd[3], *eg_at[3], *eg_bn[3];
  for (int i = 0; i < 3; ++i) {
    const int b = 5 + 6 * i;
    eg_fij[i] = (const float*)d_in[b + 0];
    eg_ni[i]  = (const float*)d_in[b + 1];
    eg_nj[i]  = (const float*)d_in[b + 2];
    eg_nd[i]  = (const float*)d_in[b + 3];
    eg_at[i]  = (const float*)d_in[b + 4];
    eg_bn[i]  = (const float*)d_in[b + 5];
  }
  const float *lW[8], *lB[8];
  for (int i = 0; i < 8; ++i) { lW[i] = (const float*)d_in[23 + 2 * i]; lB[i] = (const float*)d_in[24 + 2 * i]; }
  const float *mlW[4], *mlB[4];
  for (int i = 0; i < 4; ++i) { mlW[i] = (const float*)d_in[39 + 2 * i]; mlB[i] = (const float*)d_in[40 + 2 * i]; }

  // workspace carve (~175 MB)
  size_t off = 0;
  auto carve = [&](size_t bytes) -> char* {
    char* p = (char*)d_ws + off;
    off = (off + bytes + 255) & ~(size_t)255;
    return p;
  };
  _Float16* nf16  = (_Float16*)carve((size_t)NN * FD * 2);
  _Float16* ef16  = (_Float16*)carve((size_t)NE * FD * 2);
  _Float16* fni   = (_Float16*)carve((size_t)NN * FD * 2);
  _Float16* fnj   = (_Float16*)carve((size_t)NN * FD * 2);
  _Float16* hnode = (_Float16*)carve((size_t)NN * FD * 2);
  float*    hout  = (float*)   carve((size_t)NN * FD * 4);
  float*    elog  = (float*)   carve((size_t)NE * NH * 4);
  unsigned* menc  = (unsigned*)carve((size_t)NN * NH * 4);
  float*    ssum  = (float*)   carve((size_t)NN * NH * 4);
  _Float16* htrip = (_Float16*)carve((size_t)NT * 128 * 2);
  _Float16* hfin  = (_Float16*)carve((size_t)NT * 64 * 2);
  float*    Ub    = (float*)   carve((size_t)NT * 32 * 4);
  float*    Vb    = (float*)   carve((size_t)NT * 32 * 4);
  int*      tidx  = (int*)     carve((size_t)NT * 4);
  auto carveH = [&](size_t n) -> _Float16* { return (_Float16*)carve(n * 2); };

  const int egK[3]   = {64, 256, 256};   // node-proj K
  const int egKe[3]  = {16, 256, 256};   // edge-proj K (real)
  const int egKeP[3] = {32, 256, 256};   // edge-proj K (padded)
  _Float16 *wt_ni[3], *wt_nj[3], *wt_nd[3], *wt_fij[3];
  for (int i = 0; i < 3; ++i) {
    wt_ni[i]  = carveH((size_t)FD * egK[i]);
    wt_nj[i]  = carveH((size_t)FD * egK[i]);
    wt_nd[i]  = carveH((size_t)FD * egK[i]);
    wt_fij[i] = carveH((size_t)FD * egKeP[i]);
  }
  const int mlNC[4] = {128, 128, 128, 64};
  _Float16* wt_ml[4];
  for (int i = 0; i < 4; ++i) wt_ml[i] = carveH((size_t)mlNC[i] * 128);
  _Float16* wt_l0t = carveH(32 * 64);
  _Float16* wt_l0b = carveH(32 * 64);
  const int plK[6]  = {32, 64, 128, 128, 64, 32};
  const int plNC[6] = {64, 128, 128, 64, 32, 16};
  _Float16* wt_pl[6];
  for (int i = 0; i < 6; ++i) wt_pl[i] = carveH((size_t)plNC[i] * plK[i]);

  auto prep = [&](const float* W, _Float16* Wt, int Kr, int NC, int Kpad, int rowOff) {
    const int total = NC * Kpad;
    prep_wt_kernel<<<(total + 255) / 256, 256, 0, stream>>>(W, Wt, Kr, NC, Kpad, rowOff);
  };
  for (int i = 0; i < 3; ++i) {
    prep(eg_ni[i],  wt_ni[i],  egK[i], FD, egK[i], 0);
    prep(eg_nj[i],  wt_nj[i],  egK[i], FD, egK[i], 0);
    prep(eg_nd[i],  wt_nd[i],  egK[i], FD, egK[i], 0);
    prep(eg_fij[i], wt_fij[i], egKe[i], FD, egKeP[i], 0);
  }
  for (int i = 0; i < 4; ++i) prep(mlW[i], wt_ml[i], 128, mlNC[i], 128, 0);
  prep(lW[0], wt_l0t, 64, 32, 64, 0);
  prep(lW[0], wt_l0b, 64, 32, 64, 64);
  for (int i = 0; i < 6; ++i) prep(lW[i + 1], wt_pl[i], plK[i], plNC[i], plK[i], 0);

  const dim3 bW(128);
  const dim3 gN((NN / 16 + 3) / 4, FD / 64);     // node GEMMs: 313 x 4
  const dim3 gE(NE / 64, NH);                    // edge kernel: 3125 x 4 heads
  const int gSM = (NE * NH + 255) / 256;
  const int gMSG = (int)(((long)NE * 32 + 255) / 256);
  const int gF2H = (int)(((long)NN * FD + 255) / 256);

  auto layer_tail = [&](bool to_nf16) {
    segmax_kernel<<<gSM, 256, 0, stream>>>(elog, dst, menc, NE);
    expsum_kernel<<<gSM, 256, 0, stream>>>(elog, dst, menc, ssum, NE);
    msg_kernel<<<gMSG, 256, 0, stream>>>(hnode, elog, ssum, src, dst, hout, (long)NE * 32);
    if (to_nf16) f2h_kernel<<<gF2H, 256, 0, stream>>>(hout, nf16, (long)NN * FD);
  };
  auto zero_layer = [&]() {
    hipMemsetAsync(menc, 0, (size_t)NN * NH * 4, stream);
    hipMemsetAsync(ssum, 0, (size_t)NN * NH * 4, stream);
    hipMemsetAsync(hout, 0, (size_t)NN * FD * 4, stream);
  };

  // -------- EGAT layer 0 (f32 inputs read directly, converted in the A-loader) --------
  zero_layer();
  gemm_kernel<float, false><<<gN, bW, 0, stream>>>(nfeat, NIN, NIN, NIN, wt_ni[0], nullptr, fni, FD, NN);
  gemm_kernel<float, false><<<gN, bW, 0, stream>>>(nfeat, NIN, NIN, NIN, wt_nj[0], nullptr, fnj, FD, NN);
  gemm_kernel<float, false><<<gN, bW, 0, stream>>>(nfeat, NIN, NIN, NIN, wt_nd[0], eg_bn[0], hnode, FD, NN);
  edge_kernel<float, true, true><<<gE, bW, 0, stream>>>(efeat, EIN, EIN, 32, wt_fij[0],
      fni, fnj, src, dst, eg_at[0], ef16, elog, NE);
  layer_tail(true);

  // -------- EGAT layer 1 (ef16 updated in place: racy, the 4 head blocks share rows) --------
  zero_layer();
  gemm_kernel<_Float16, false><<<gN, bW, 0, stream>>>(nf16, FD, FD, FD, wt_ni[1], nullptr, fni, FD, NN);
  gemm_kernel<_Float16, false><<<gN, bW, 0, stream>>>(nf16, FD, FD, FD, wt_nj[1], nullptr, fnj, FD, NN);
  gemm_kernel<_Float16, false><<<gN, bW, 0, stream>>>(nf16, FD, FD, FD, wt_nd[1], eg_bn[1], hnode, FD, NN);
  edge_kernel<_Float16, false, true><<<gE, bW, 0, stream>>>(ef16, FD, FD, FD, wt_fij[1],
      fni, fnj, src, dst, eg_at[1], ef16, elog, NE);
  layer_tail(true);

  // -------- EGAT layer 2 (edge output discarded -> no store) --------
  zero_layer();
  gemm_kernel<_Float16, false><<<gN, bW, 0, stream>>>(nf16, FD, FD, FD, wt_ni[2], nullptr, fni, FD, NN);
  gemm_kernel<_Float16, false><<<gN, bW, 0, stream>>>(nf16, FD, FD, FD, wt_nj[2], nullptr, fnj, FD, NN);
  gemm_kernel<_Float16, false><<<gN, bW, 0, stream>>>(nf16, FD, FD, FD, wt_nd[2], eg_bn[2], hnode, FD, NN);
  edge_kernel<_Float16, false, false><<<gE, bW, 0, stream>>>(ef16, FD, FD, FD, wt_fij[2],
      fni, fnj, src, dst, eg_at[2], (_Float16*)nullptr, elog, NE);
  layer_tail(false);

  // -------- head --------
  trip_kernel<<<1, 1, 0, stream>>>(mask, tidx, NN, NT);
  gather_kernel<<<(NT * 128 + 255) / 256, 256, 0, stream>>>(hout, nfeat, tidx, htrip);
  ml_kernel<<<8, 128, 0, stream>>>(htrip, wt_ml[0], mlB[0], wt_ml[1], mlB[1],
                                   wt_ml[2], mlB[2], wt_ml[3], mlB[3], hfin);
  uv_kernel<<<8, 128, 0, stream>>>(hfin, wt_l0t, wt_l0b, Ub, Vb);
  pairs_kernel<<<NPAIR / 64, 128, 0, stream>>>(Ub, Vb, lB[0],
      wt_pl[0], lB[1], wt_pl[1], lB[2], wt_pl[2], lB[3],
      wt_pl[3], lB[4], wt_pl[4], lB[5], wt_pl[5], lB[6],
      lW[7], lB[7], (float*)d_out);
}
